// CombinedModel_8529805049887
// MI455X (gfx1250) — compile-verified, matrix operand order checked by reading, not run on hardware
//
#include <hip/hip_runtime.h>
#include <hip/hip_bf16.h>

// ---------------------------------------------------------------------------
// CDNA5 (gfx1250) implementation: all dense GEMMs through
// v_wmma_f32_16x16x32_bf16 (wave32), scatter phase via global_atomic_add_f32.
// GEMM inner loop is explicitly double-buffered so the scheduler can overlap
// L2 load latency with WMMA issue (partial s_wait_loadcnt instead of 0).
// ---------------------------------------------------------------------------

#define N_NODES 50000
#define N_EDGES 800000
#define GNN_IN  256
#define GNN_HID 64
#define COMB    64
#define MLP_IN  256
#define MLP_HID 512

typedef __attribute__((ext_vector_type(16))) __bf16 v16bf;
typedef __attribute__((ext_vector_type(8)))  float  v8f;

union V16 {
    uint4 u[2];
    v16bf v;
};

__device__ __forceinline__ unsigned short f2bf(float f) {
    unsigned int u = __float_as_uint(f);
    unsigned int r = u + 0x7FFFu + ((u >> 16) & 1u);   // round-to-nearest-even
    return (unsigned short)(r >> 16);
}

// ---------------------------------------------------------------------------
// fp32 -> bf16 streaming conversion (4 elems / thread)
// ---------------------------------------------------------------------------
__global__ void cvt_bf16_kernel(const float* __restrict__ in,
                                unsigned short* __restrict__ out, int n4) {
    int i = blockIdx.x * blockDim.x + threadIdx.x;
    if (i >= n4) return;
    const float4 f = ((const float4*)in)[i];
    ushort4 o;
    o.x = f2bf(f.x); o.y = f2bf(f.y); o.z = f2bf(f.z); o.w = f2bf(f.w);
    ((ushort4*)out)[i] = o;
}

// ---------------------------------------------------------------------------
// Pack weight W [K x N, row-major f32] into per-lane WMMA B-operand tiles.
// Tile = 32(K) x 16(N). Per lane: 16 bf16 contiguous (32 B). Lane L<16 owns
// column nt*16+L with K {kb..kb+7, kb+16..kb+23} (kb = kt*32); lanes 16-31
// own the K {8..15, 24..31} halves, mirroring the A-operand layout so the
// GEMM inner loop is two b128 loads per operand.
// ---------------------------------------------------------------------------
__global__ void pack_w_kernel(const float* __restrict__ W,
                              unsigned short* __restrict__ P, int K, int N) {
    int idx = blockIdx.x * blockDim.x + threadIdx.x;
    int nTiles = N >> 4;
    int total = (K >> 5) * nTiles * 32;
    if (idx >= total) return;
    int lane = idx & 31;
    int tile = idx >> 5;
    int kt = tile / nTiles;
    int nt = tile % nTiles;
    int col = nt * 16 + (lane & 15);
    int kb  = kt * 32 + ((lane < 16) ? 0 : 8);
    __attribute__((aligned(16))) unsigned short o[16];
#pragma unroll
    for (int j = 0; j < 8; ++j) o[j]     = f2bf(W[(size_t)(kb + j) * N + col]);
#pragma unroll
    for (int j = 0; j < 8; ++j) o[8 + j] = f2bf(W[(size_t)(kb + 16 + j) * N + col]);
    uint4* dst = (uint4*)(P + (size_t)tile * 512 + lane * 16);
    dst[0] = ((uint4*)o)[0];
    dst[1] = ((uint4*)o)[1];
}

// ---------------------------------------------------------------------------
// WMMA GEMM: C[M,N] = act(A[M,K](bf16) * B(packed bf16) + bias)
// 8 waves / block; each wave owns a 16x64 output slab (4 wmma accumulators,
// 4x A-tile reuse). Double-buffered K loop: tile kt+1 operands are in flight
// while the four WMMAs consume tile kt. All control flow is wave-uniform ->
// EXEC is all-ones at every WMMA as the ISA requires.
// ---------------------------------------------------------------------------
__global__ void gemm_wmma_kernel(const unsigned short* __restrict__ A,
                                 const unsigned short* __restrict__ Bp,
                                 const float* __restrict__ bias,
                                 void* __restrict__ C,
                                 int M, int N, int K, int doRelu, int outBf16) {
    const int lane = threadIdx.x & 31;
    const int wid  = threadIdx.x >> 5;
    const int nGroups = N >> 6;     // 64 output cols per wave
    const int mTiles  = M >> 4;
    const int task = blockIdx.x * 8 + wid;
    if (task >= mTiles * nGroups) return;   // whole-wave uniform exit
    const int mt = task / nGroups;
    const int ng = task % nGroups;
    const int kTiles = K >> 5;
    const int nTiles = N >> 4;

    const int hsel = (lane < 16) ? 0 : 8;   // A half-lane K-offset per ISA layout
    const unsigned short* arow = A + (size_t)(mt * 16 + (lane & 15)) * K;
    const unsigned short* bcol = Bp + (size_t)(ng << 2) * 512 + lane * 16;
    const size_t bstride = (size_t)nTiles * 512;   // halves per K-tile row

    v8f acc0 = {}, acc1 = {}, acc2 = {}, acc3 = {};

    V16 aC, aN;
    V16 bC0, bC1, bC2, bC3, bN0, bN1, bN2, bN3;

    // prologue: tile 0
    {
        const uint4* ap = (const uint4*)(arow + hsel);
        aC.u[0] = ap[0];
        aC.u[1] = ap[2];
        const unsigned short* bt = bcol;
        bC0.u[0] = ((const uint4*)(bt + 0 * 512))[0];
        bC0.u[1] = ((const uint4*)(bt + 0 * 512))[1];
        bC1.u[0] = ((const uint4*)(bt + 1 * 512))[0];
        bC1.u[1] = ((const uint4*)(bt + 1 * 512))[1];
        bC2.u[0] = ((const uint4*)(bt + 2 * 512))[0];
        bC2.u[1] = ((const uint4*)(bt + 2 * 512))[1];
        bC3.u[0] = ((const uint4*)(bt + 3 * 512))[0];
        bC3.u[1] = ((const uint4*)(bt + 3 * 512))[1];
    }

    for (int kt = 0; kt < kTiles; ++kt) {
        // issue next tile's loads (clamped on last iteration; branch-free)
        const int ktn = (kt + 1 < kTiles) ? (kt + 1) : kt;
        {
            const uint4* ap = (const uint4*)(arow + ktn * 32 + hsel);
            aN.u[0] = ap[0];
            aN.u[1] = ap[2];
            const unsigned short* bt = bcol + (size_t)ktn * bstride;
            bN0.u[0] = ((const uint4*)(bt + 0 * 512))[0];
            bN0.u[1] = ((const uint4*)(bt + 0 * 512))[1];
            bN1.u[0] = ((const uint4*)(bt + 1 * 512))[0];
            bN1.u[1] = ((const uint4*)(bt + 1 * 512))[1];
            bN2.u[0] = ((const uint4*)(bt + 2 * 512))[0];
            bN2.u[1] = ((const uint4*)(bt + 2 * 512))[1];
            bN3.u[0] = ((const uint4*)(bt + 3 * 512))[0];
            bN3.u[1] = ((const uint4*)(bt + 3 * 512))[1];
        }

        acc0 = __builtin_amdgcn_wmma_f32_16x16x32_bf16(false, aC.v, false, bC0.v,
                                                       (short)0, acc0, false, false);
        acc1 = __builtin_amdgcn_wmma_f32_16x16x32_bf16(false, aC.v, false, bC1.v,
                                                       (short)0, acc1, false, false);
        acc2 = __builtin_amdgcn_wmma_f32_16x16x32_bf16(false, aC.v, false, bC2.v,
                                                       (short)0, acc2, false, false);
        acc3 = __builtin_amdgcn_wmma_f32_16x16x32_bf16(false, aC.v, false, bC3.v,
                                                       (short)0, acc3, false, false);

        aC = aN;
        bC0 = bN0; bC1 = bN1; bC2 = bN2; bC3 = bN3;
    }

    // C/D 16x16 f32 layout: VGPR v, lanes 0-15 -> M=v, lanes 16-31 -> M=v+8.
    const int r0 = mt * 16 + ((lane < 16) ? 0 : 8);
    v8f accs[4] = {acc0, acc1, acc2, acc3};
#pragma unroll
    for (int j = 0; j < 4; ++j) {
        const int cc = (ng * 4 + j) * 16 + (lane & 15);
        const float bv = bias ? bias[cc] : 0.0f;
#pragma unroll
        for (int v = 0; v < 8; ++v) {
            float val = accs[j][v] + bv;
            if (doRelu) val = fmaxf(val, 0.0f);
            const size_t off = (size_t)(r0 + v) * N + cc;
            if (outBf16) ((unsigned short*)C)[off] = f2bf(val);
            else         ((float*)C)[off] = val;
        }
    }
}

// ---------------------------------------------------------------------------
// Degree / normalization helpers (GCN sym-norm with self-loops, weight 1)
// ---------------------------------------------------------------------------
__global__ void deg_init_kernel(float* __restrict__ deg, int n) {
    int i = blockIdx.x * blockDim.x + threadIdx.x;
    if (i < n) deg[i] = 1.0f;   // self-loop weight
}

__global__ void deg_accum_kernel(const int* __restrict__ dst,
                                 const float* __restrict__ ew,
                                 float* __restrict__ deg, int nE) {
    int e = blockIdx.x * blockDim.x + threadIdx.x;
    if (e < nE) atomicAdd(&deg[dst[e]], ew[e]);
}

__global__ void dinv_kernel(const float* __restrict__ deg,
                            float* __restrict__ dinv, int n) {
    int i = blockIdx.x * blockDim.x + threadIdx.x;
    if (i >= n) return;
    float d = deg[i];
    dinv[i] = (d > 0.0f) ? rsqrtf(fmaxf(d, 1e-12f)) : 0.0f;
}

// norm[e] = dinv[src]*ew*dinv[dst]  (computed once, reused by both GCN layers)
__global__ void edge_norm_kernel(const int* __restrict__ src,
                                 const int* __restrict__ dst,
                                 const float* __restrict__ ew,
                                 const float* __restrict__ dinv,
                                 float* __restrict__ norm, int nE) {
    int e = blockIdx.x * blockDim.x + threadIdx.x;
    if (e < nE) norm[e] = dinv[src[e]] * ew[e] * dinv[dst[e]];
}

// agg[n][:] = h[n][:] * dinv[n]^2   (self-loop term; also zero-initializes agg)
__global__ void agg_init_kernel(const float* __restrict__ h,
                                const float* __restrict__ dinv,
                                float* __restrict__ agg, int nNodes) {
    int idx = blockIdx.x * blockDim.x + threadIdx.x;
    int n = idx >> 4;
    if (n >= nNodes) return;
    int c = (idx & 15) * 4;
    float w = dinv[n]; w *= w;
    float4 hv = *(const float4*)(h + (size_t)n * COMB + c);
    float4 o = {hv.x * w, hv.y * w, hv.z * w, hv.w * w};
    *(float4*)(agg + (size_t)n * COMB + c) = o;
}

// agg[dst] += h[src] * norm[e]; 16 threads/edge, float4 each
__global__ void gcn_scatter_kernel(const float* __restrict__ h,
                                   const int* __restrict__ src,
                                   const int* __restrict__ dst,
                                   const float* __restrict__ norm,
                                   float* __restrict__ agg, int nE) {
    int idx = blockIdx.x * blockDim.x + threadIdx.x;
    int e = idx >> 4;
    if (e >= nE) return;
    int c = (idx & 15) * 4;
    int s = src[e], d = dst[e];
    float nm = norm[e];
    float4 hv = *(const float4*)(h + (size_t)s * COMB + c);
    float* ap = agg + (size_t)d * COMB + c;
    atomicAdd(ap + 0, hv.x * nm);
    atomicAdd(ap + 1, hv.y * nm);
    atomicAdd(ap + 2, hv.z * nm);
    atomicAdd(ap + 3, hv.w * nm);
}

// g = relu(agg + b), emitted as bf16 (A operand of next WMMA GEMM)
__global__ void relu_bias_bf16_kernel(const float* __restrict__ agg,
                                      const float* __restrict__ b,
                                      unsigned short* __restrict__ g, int total) {
    int idx = blockIdx.x * blockDim.x + threadIdx.x;
    if (idx >= total) return;
    float v = agg[idx] + b[idx & (COMB - 1)];
    g[idx] = f2bf(fmaxf(v, 0.0f));
}

// out[n] = [mlp_out[n] ; agg2[n]+b2] . fcf_W + fcf_b  (128-wide dot / block)
__global__ void final_kernel(const float* __restrict__ mlp,
                             const float* __restrict__ agg2,
                             const float* __restrict__ b2,
                             const float* __restrict__ fw,
                             const float* __restrict__ fb,
                             float* __restrict__ out) {
    __shared__ float red[128];
    int n = blockIdx.x;
    int t = threadIdx.x;
    float v;
    if (t < COMB) v = mlp[(size_t)n * COMB + t];
    else          v = agg2[(size_t)n * COMB + (t - COMB)] + b2[t - COMB];
    red[t] = v * fw[t];
    __syncthreads();
    for (int off = 64; off > 0; off >>= 1) {
        if (t < off) red[t] += red[t + off];
        __syncthreads();
    }
    if (t == 0) out[n] = red[0] + fb[0];
}

// ---------------------------------------------------------------------------
extern "C" void kernel_launch(void* const* d_in, const int* in_sizes, int n_in,
                              void* d_out, int out_size, void* d_ws, size_t ws_size,
                              hipStream_t stream) {
    (void)in_sizes; (void)n_in; (void)out_size; (void)ws_size;

    const float* x        = (const float*)d_in[0];
    const int*   eidx     = (const int*)d_in[1];
    const float* eattr    = (const float*)d_in[2];
    const float* ufeat    = (const float*)d_in[3];
    const float* gcn1_W   = (const float*)d_in[4];
    const float* gcn1_b   = (const float*)d_in[5];
    const float* gcn2_W   = (const float*)d_in[6];
    const float* gcn2_b   = (const float*)d_in[7];
    const float* fc1_W    = (const float*)d_in[8];
    const float* fc1_b    = (const float*)d_in[9];
    const float* fc2_W    = (const float*)d_in[10];
    const float* fc2_b    = (const float*)d_in[11];
    const float* fcf_W    = (const float*)d_in[12];
    const float* fcf_b    = (const float*)d_in[13];
    const int* esrc = eidx;
    const int* edst = eidx + N_EDGES;
    float* out = (float*)d_out;

    // ---- workspace carve --------------------------------------------------
    char* ws = (char*)d_ws;
    size_t off = 0;
    auto carve = [&](size_t bytes) -> void* {
        void* p = ws + off;
        off = (off + bytes + 255) & ~(size_t)255;
        return p;
    };
    unsigned short* UBF = (unsigned short*)carve((size_t)N_NODES * MLP_IN * 2);
    unsigned short* XBF = (unsigned short*)carve((size_t)N_NODES * GNN_IN * 2);
    unsigned short* HBF = (unsigned short*)carve((size_t)N_NODES * MLP_HID * 2);
    unsigned short* GBF = (unsigned short*)carve((size_t)N_NODES * COMB * 2);
    unsigned short* W1P = (unsigned short*)carve((size_t)MLP_IN * MLP_HID * 2);
    unsigned short* W2P = (unsigned short*)carve((size_t)MLP_HID * COMB * 2);
    unsigned short* G1P = (unsigned short*)carve((size_t)GNN_IN * GNN_HID * 2);
    unsigned short* G2P = (unsigned short*)carve((size_t)GNN_HID * COMB * 2);
    float* MLPOUT = (float*)carve((size_t)N_NODES * COMB * 4);
    float* H1     = (float*)carve((size_t)N_NODES * COMB * 4);  // reused for H2
    float* AGG    = (float*)carve((size_t)N_NODES * COMB * 4);  // agg1 then agg2
    float* DEG    = (float*)carve((size_t)N_NODES * 4);
    float* DINV   = (float*)carve((size_t)N_NODES * 4);
    float* NORM   = (float*)carve((size_t)N_EDGES * 4);
    float* H2 = H1;

    // ---- activation conversion + weight packing ---------------------------
    {
        int n4 = (N_NODES * MLP_IN) / 4;
        cvt_bf16_kernel<<<(n4 + 255) / 256, 256, 0, stream>>>(ufeat, UBF, n4);
        n4 = (N_NODES * GNN_IN) / 4;
        cvt_bf16_kernel<<<(n4 + 255) / 256, 256, 0, stream>>>(x, XBF, n4);
    }
    auto pack = [&](const float* W, unsigned short* P, int K, int N) {
        int total = (K / 32) * (N / 16) * 32;
        pack_w_kernel<<<(total + 255) / 256, 256, 0, stream>>>(W, P, K, N);
    };
    pack(fc1_W, W1P, MLP_IN, MLP_HID);
    pack(fc2_W, W2P, MLP_HID, COMB);
    pack(gcn1_W, G1P, GNN_IN, GNN_HID);
    pack(gcn2_W, G2P, GNN_HID, COMB);

    // ---- degree / sym-norm / per-edge norm --------------------------------
    deg_init_kernel<<<(N_NODES + 255) / 256, 256, 0, stream>>>(DEG, N_NODES);
    deg_accum_kernel<<<(N_EDGES + 255) / 256, 256, 0, stream>>>(edst, eattr, DEG, N_EDGES);
    dinv_kernel<<<(N_NODES + 255) / 256, 256, 0, stream>>>(DEG, DINV, N_NODES);
    edge_norm_kernel<<<(N_EDGES + 255) / 256, 256, 0, stream>>>(esrc, edst, eattr, DINV,
                                                                NORM, N_EDGES);

    auto gemm = [&](const unsigned short* A, const unsigned short* Bp,
                    const float* bias, void* C, int M, int N, int K,
                    int relu, int outBf) {
        int tasks = (M / 16) * (N / 64);
        int blocks = (tasks + 7) / 8;
        gemm_wmma_kernel<<<blocks, 256, 0, stream>>>(A, Bp, bias, C, M, N, K, relu, outBf);
    };

    // ---- MLP branch: H = relu(U@W1+b1) (bf16) ; mlp_out = H@W2+b2 (f32) ---
    gemm(UBF, W1P, fc1_b, HBF, N_NODES, MLP_HID, MLP_IN, /*relu=*/1, /*bf16=*/1);
    gemm(HBF, W2P, fc2_b, MLPOUT, N_NODES, COMB, MLP_HID, 0, 0);

    // ---- GCN layer 1 -------------------------------------------------------
    gemm(XBF, G1P, nullptr, H1, N_NODES, GNN_HID, GNN_IN, 0, 0);
    agg_init_kernel<<<(N_NODES * 16 + 255) / 256, 256, 0, stream>>>(H1, DINV, AGG, N_NODES);
    gcn_scatter_kernel<<<(N_EDGES * 16 + 255) / 256, 256, 0, stream>>>(
        H1, esrc, edst, NORM, AGG, N_EDGES);
    relu_bias_bf16_kernel<<<(N_NODES * COMB + 255) / 256, 256, 0, stream>>>(
        AGG, gcn1_b, GBF, N_NODES * COMB);

    // ---- GCN layer 2 (reuse H1/AGG buffers) -------------------------------
    gemm(GBF, G2P, nullptr, H2, N_NODES, COMB, GNN_HID, 0, 0);
    agg_init_kernel<<<(N_NODES * 16 + 255) / 256, 256, 0, stream>>>(H2, DINV, AGG, N_NODES);
    gcn_scatter_kernel<<<(N_EDGES * 16 + 255) / 256, 256, 0, stream>>>(
        H2, esrc, edst, NORM, AGG, N_EDGES);

    // ---- final: concat + 128-dot + bias -----------------------------------
    final_kernel<<<N_NODES, 128, 0, stream>>>(MLPOUT, AGG, gcn2_b, fcf_W, fcf_b, out);
}
